// GraphAttentionLayer_14860586844392
// MI455X (gfx1250) — matrix operand order repaired and checked by reading, not compiled, not run on hardware
//
#include <hip/hip_runtime.h>
#include <hip/hip_bf16.h>
#include <math.h>

// Problem constants (from reference)
constexpr int Bn = 4, LQ = 1024, LK = 1024, Dm = 512, Hh = 8, HD = 64;
constexpr float NEG_SLOPE = 0.2f;

typedef __attribute__((ext_vector_type(16))) __bf16        v16bf;
typedef __attribute__((ext_vector_type(2)))  __bf16        v2bf;
typedef __attribute__((ext_vector_type(8)))  float         v8f;
typedef __attribute__((ext_vector_type(8)))  unsigned int  v8u;
typedef int v4i_t __attribute__((vector_size(16)));
typedef unsigned short bf16raw;

// ---------------- feature detection (compile-safe) -------------------------
#if defined(__has_builtin)
#  if __has_builtin(__builtin_amdgcn_cvt_pk_bf16_f32)
#    define HAVE_PK_CVT 1
#  endif
#  if __has_builtin(__builtin_amdgcn_global_load_async_to_lds_b128)
#    define HAVE_ASYNC 1
#  endif
#endif
#ifndef HAVE_PK_CVT
#  define HAVE_PK_CVT 0
#endif
#ifndef HAVE_ASYNC
#  define HAVE_ASYNC 0
#endif

__device__ __forceinline__ unsigned int f2bf(float x) {   // RNE, bit-level
    unsigned int u = __float_as_uint(x);
    u = (u + 0x7FFFu + ((u >> 16) & 1u)) >> 16;
    return u & 0xFFFFu;
}

__device__ __forceinline__ unsigned int packpair(float lo, float hi) {
#if HAVE_PK_CVT
    v2bf r = __builtin_amdgcn_cvt_pk_bf16_f32(lo, hi);
    return __builtin_bit_cast(unsigned int, r);
#else
    return f2bf(lo) | (f2bf(hi) << 16);
#endif
}

__device__ __forceinline__ v16bf pack16(const float* f) {
    v8u u;
#pragma unroll
    for (int i = 0; i < 8; ++i) u[i] = packpair(f[2 * i], f[2 * i + 1]);
    return __builtin_bit_cast(v16bf, u);
}

struct U32x8 { uint4 lo, hi; };
// load 16 bf16 (two 16B segments) straight into a WMMA operand
__device__ __forceinline__ v16bf load_v16bf(const bf16raw* p0, const bf16raw* p1) {
    U32x8 t;
    t.lo = *(const uint4*)p0;
    t.hi = *(const uint4*)p1;
    return __builtin_bit_cast(v16bf, t);
}

#if HAVE_ASYNC
// builtin signature: (v4i addrspace(1)*, v4i addrspace(3)*, imm offset, imm cpol)
#define ASYNC_B128(gsrc, ldst)                                              \
    __builtin_amdgcn_global_load_async_to_lds_b128(                         \
        (__attribute__((address_space(1))) v4i_t*)(gsrc),                   \
        (__attribute__((address_space(3))) v4i_t*)(ldst), 0, 0)
#  if defined(__has_builtin) && __has_builtin(__builtin_amdgcn_s_wait_asynccnt)
#    define ASYNC_WAIT(n) __builtin_amdgcn_s_wait_asynccnt(n)
#  else
#    define ASYNC_WAIT(n) asm volatile("s_wait_asynccnt " #n ::: "memory")
#  endif
#else
#  define ASYNC_WAIT(n)
#endif

// ---------------------------------------------------------------------------
// Kernel 0: elementwise f32 -> bf16 (n multiple of 4)
// ---------------------------------------------------------------------------
__global__ __launch_bounds__(256) void cvt_kernel(const float* __restrict__ src,
                                                  bf16raw* __restrict__ dst, int n) {
    const int i = (blockIdx.x * 256 + threadIdx.x) * 4;
    if (i < n) {
        const float4 v = *(const float4*)(src + i);
        uint2 r;
        r.x = packpair(v.x, v.y);
        r.y = packpair(v.z, v.w);
        *(uint2*)(dst + i) = r;
    }
}

// ---------------------------------------------------------------------------
// Kernel 1: Y[m,n] = sum_k Xb[m,k] * Wb[n,k]  (bf16 in, f32 out)
// 8 waves; wave: 16 rows x 64 cols; WG: 128 x 64.
// W tile double-buffered in LDS via async-to-LDS (1 async/thread/chunk).
// ---------------------------------------------------------------------------
__global__ __launch_bounds__(256) void proj_kernel(const bf16raw* __restrict__ Xb,
                                                   const bf16raw* __restrict__ Wb,
                                                   float* __restrict__ Y) {
    __shared__ bf16raw wb_s[2][64 * 40];         // 2 x (64 cols x 32 k), pad-40

    const int wv   = threadIdx.x >> 5;
    const int lane = threadIdx.x & 31;
    const int l16  = lane & 15;
    const bool hi  = lane >= 16;

    const int rowbase = blockIdx.x * 128 + wv * 16;
    const int row     = rowbase + l16;
    const int colbase = blockIdx.y * 64;

    const int kb    = hi ? 8 : 0;                // A K base
    const int khalf = hi ? 16 : 0;               // unused: B also uses kb

    v8f acc[4];
#pragma unroll
    for (int t = 0; t < 4; ++t)
#pragma unroll
        for (int j = 0; j < 8; ++j) acc[t][j] = 0.0f;

    const bf16raw* xrow = Xb + (size_t)row * Dm;
    // staging coords: thread -> (col row, 8-elem segment)
    const int srow = threadIdx.x >> 2;           // 0..63
    const int sseg = (threadIdx.x & 3) * 8;      // 0,8,16,24

    auto stage_w = [&](int k0c, int p) {
        const bf16raw* wsrc = Wb + (size_t)(colbase + srow) * Dm + k0c + sseg;
        bf16raw* wdst = &wb_s[p][srow * 40 + sseg];
#if HAVE_ASYNC
        ASYNC_B128(wsrc, wdst);
#else
        *(uint4*)wdst = *(const uint4*)wsrc;
#endif
    };

    stage_w(0, 0);
    for (int k0 = 0; k0 < Dm; k0 += 32) {
        const int p = (k0 >> 5) & 1;
        if (k0 + 32 < Dm) { stage_w(k0 + 32, p ^ 1); ASYNC_WAIT(1); }
        else              { ASYNC_WAIT(0); }
        __syncthreads();

        const v16bf A = load_v16bf(xrow + k0 + kb, xrow + k0 + kb + 16);
#pragma unroll
        for (int t = 0; t < 4; ++t) {
            const int nloc = t * 16 + l16;
            const v16bf Bm = load_v16bf(&wb_s[p][nloc * 40 + kb],
                                        &wb_s[p][nloc * 40 + kb + 16]);
            acc[t] = __builtin_amdgcn_wmma_f32_16x16x32_bf16(
                false, A, false, Bm, (short)0, acc[t], false, false);
        }
        __syncthreads();
    }

#pragma unroll
    for (int t = 0; t < 4; ++t) {
        const int col = colbase + t * 16 + l16;
#pragma unroll
        for (int j = 0; j < 8; ++j) {
            const int r = rowbase + j + (hi ? 8 : 0);
            Y[(size_t)r * Dm + col] = acc[t][j];
        }
    }
}

// ---------------------------------------------------------------------------
// Kernel 2: sq[b,h,q] = Q[b,q,h,:]·a_q[h,:] ; sk[b,h,k] = K[b,k,h,:]·a_k[h,:]
// ---------------------------------------------------------------------------
__global__ __launch_bounds__(256) void sqsk_kernel(const float* __restrict__ Qp,
                                                   const float* __restrict__ Kp,
                                                   const float* __restrict__ a,
                                                   float* __restrict__ sq,
                                                   float* __restrict__ sk) {
    const int total = Bn * LQ * Hh;
    int idx = blockIdx.x * blockDim.x + threadIdx.x;
    const bool isK = idx >= total;
    int i = isK ? idx - total : idx;
    const int h  = i % Hh;
    const int bl = i / Hh;
    const float* P  = (isK ? Kp : Qp) + (size_t)bl * Dm + h * HD;
    const float* av = a + h * (2 * HD) + (isK ? HD : 0);
    float s = 0.0f;
#pragma unroll 8
    for (int d = 0; d < HD; ++d) s += P[d] * av[d];
    const int b = bl / LQ, l = bl % LQ;
    (isK ? sk : sq)[((size_t)b * Hh + h) * LQ + l] = s;
}

// ---------------------------------------------------------------------------
// Kernel 3a: per row (b,h,q): m = max_k masked f(sq+sk), Z = sum exp(s-m)
// One wave32 per row, online softmax stats + wave reduction.
// ---------------------------------------------------------------------------
__global__ __launch_bounds__(256) void stats_kernel(const float* __restrict__ sq,
                                                    const float* __restrict__ sk,
                                                    const int* __restrict__ mask,
                                                    float* __restrict__ mrow,
                                                    float* __restrict__ zrow) {
    const int wv   = threadIdx.x >> 5;
    const int lane = threadIdx.x & 31;
    const int gw   = blockIdx.x * 8 + wv;        // (b*H + h)*LQ + q
    const int q    = gw % LQ;
    const int bh   = gw / LQ;
    const int b    = bh / Hh;

    const float sqv = sq[gw];
    const float* skp = sk + (size_t)bh * LK;
    const int*  mp  = mask + ((size_t)b * LQ + q) * LK;
    const float NINF = -__builtin_inff();

    float m = NINF, z = 0.0f;
    for (int k = lane; k < LK; k += 32) {
        if (mp[k] == 0) continue;
        float s = sqv + skp[k];
        s = (s >= 0.0f) ? s : NEG_SLOPE * s;
        if (s > m) { z = z * __expf(m - s) + 1.0f; m = s; }
        else       { z += __expf(s - m); }
    }
#pragma unroll
    for (int off = 16; off > 0; off >>= 1) {
        const float mo = __shfl_xor(m, off);
        const float zo = __shfl_xor(z, off);
        const float mn = fmaxf(m, mo);
        const float e1 = (m  > NINF) ? __expf(m  - mn) : 0.0f;
        const float e2 = (mo > NINF) ? __expf(mo - mn) : 0.0f;
        z = z * e1 + zo * e2;
        m = mn;
    }
    if (lane == 0) { mrow[gw] = m; zrow[gw] = z; }
}

// ---------------------------------------------------------------------------
// Kernel 3.5: transpose+convert V: Vt[b,h,d,k] (bf16) from Vp[b,k,h*64+d] (f32)
// ---------------------------------------------------------------------------
__global__ __launch_bounds__(256) void vtrans_kernel(const float* __restrict__ Vp,
                                                     bf16raw* __restrict__ Vtg) {
    __shared__ bf16raw tile[64 * 36];
    const int bh = blockIdx.x, b = bh / Hh, h = bh % Hh;
    const int k0 = blockIdx.y * 32;

    for (int i = threadIdx.x; i < 32 * 64; i += 256) {
        const int kk = i >> 6, dd = i & 63;
        const float v = Vp[((size_t)b * LK + k0 + kk) * Dm + h * HD + dd];
        tile[dd * 36 + kk] = (bf16raw)f2bf(v);
    }
    __syncthreads();
    for (int i = threadIdx.x; i < 32 * 64; i += 256) {
        const int dd = i >> 5, kk = i & 31;
        Vtg[((size_t)bh * HD + dd) * LK + k0 + kk] = tile[dd * 36 + kk];
    }
}

// ---------------------------------------------------------------------------
// Kernel 3b: out = softmax-weighted V, flash style, WMMA bf16.
// Block = 8 waves; wave = 16 q x 64 d; WG = 128 q rows, one (b,h).
// V + mask chunks double-buffered in LDS via async-to-LDS (5 async/thr/chunk).
// ---------------------------------------------------------------------------
__global__ __launch_bounds__(256) void attn_kernel(const bf16raw* __restrict__ Vtg,
                                                   const float* __restrict__ sq,
                                                   const float* __restrict__ sk,
                                                   const int* __restrict__ mask,
                                                   const float* __restrict__ mrow,
                                                   const float* __restrict__ zrow,
                                                   float* __restrict__ out) {
    __shared__ float   sk_s[LK];                 //  4 KB
    __shared__ bf16raw vt_s[2][64 * 40];         // 10 KB: V chunk [d][k] bf16, pad-40
    __shared__ int     mk_s[2][128 * 36];        // 37 KB: mask chunk [row][k], pad-36

    const int bh = blockIdx.x;
    const int b  = bh / Hh;
    const int h  = bh % Hh;
    const int wv   = threadIdx.x >> 5;
    const int lane = threadIdx.x & 31;
    const int l16  = lane & 15;
    const bool hi  = lane >= 16;
    const int qbase = blockIdx.y * 128;

    const int wrow = qbase + wv * 16;
    const int row  = wrow + l16;
    const int rowg = bh * LQ + row;
    const float sqv = sq[rowg];
    const float mv  = mrow[rowg];

    const int kb    = hi ? 8 : 0;
    const int khalf = hi ? 16 : 0;
    const bf16raw* vtrow = Vtg + (size_t)bh * HD * LK;          // + d*LK + k
    const int*    mbase  = mask + (size_t)b * LQ * LK;          // + q*LK + k
    const int lrow = wv * 16 + l16;

    // staging coords
    const int vrow = threadIdx.x >> 2, vseg = (threadIdx.x & 3) * 8;   // V: 64 x 4

    auto stage_chunk = [&](int k0c, int p) {
        // V chunk (bf16, already [d][k]): 64 rows x 64B -> 1 async/thread
        {
            const bf16raw* g = vtrow + (size_t)vrow * LK + k0c + vseg;
            bf16raw* l = &vt_s[p][vrow * 40 + vseg];
#if HAVE_ASYNC
            ASYNC_B128(g, l);
#else
            *(uint4*)l = *(const uint4*)g;
#endif
        }
        // mask chunk: 128 rows x 128B -> 4 async/thread
#pragma unroll
        for (int j = 0; j < 4; ++j) {
            const int idx = threadIdx.x + j * 256;
            const int rr = idx >> 3, seg = (idx & 7) * 4;
            const int* g = mbase + (size_t)(qbase + rr) * LK + k0c + seg;
            int* l = &mk_s[p][rr * 36 + seg];
#if HAVE_ASYNC
            ASYNC_B128(g, l);
#else
            *(int4*)l = *(const int4*)g;
#endif
        }
    };

    // stage full sk row once (completes, in order, before first ASYNC_WAIT(5))
    {
        const float* g = sk + (size_t)bh * LK + threadIdx.x * 4;
        float* l = &sk_s[threadIdx.x * 4];
#if HAVE_ASYNC
        ASYNC_B128(g, l);
#else
        *(float4*)l = *(const float4*)g;
#endif
    }
    stage_chunk(0, 0);

    v8f acc[4];
#pragma unroll
    for (int t = 0; t < 4; ++t)
#pragma unroll
        for (int j = 0; j < 8; ++j) acc[t][j] = 0.0f;

    for (int k0 = 0; k0 < LK; k0 += 32) {
        const int p = (k0 >> 5) & 1;
        if (k0 + 32 < LK) {
            stage_chunk(k0 + 32, p ^ 1);
            if (threadIdx.x < 64)
                __builtin_prefetch(vtrow + (size_t)threadIdx.x * LK + k0 + 64, 0, 1);
            ASYNC_WAIT(5);
        } else {
            ASYNC_WAIT(0);
        }
        __syncthreads();

        // ---- A tile: p = exp(leakyrelu(sq+sk) - m), masked ----
        float pf[16];
        const float4 skA = *(const float4*)&sk_s[k0 + kb];
        const float4 skB = *(const float4*)&sk_s[k0 + kb + 4];
        const float4 skC = *(const float4*)&sk_s[k0 + kb + 16];
        const float4 skD = *(const float4*)&sk_s[k0 + kb + 20];
        const int4 mA = *(const int4*)&mk_s[p][lrow * 36 + kb];
        const int4 mB = *(const int4*)&mk_s[p][lrow * 36 + kb + 4];
        const int4 mC = *(const int4*)&mk_s[p][lrow * 36 + kb + 16];
        const int4 mD = *(const int4*)&mk_s[p][lrow * 36 + kb + 20];
        const float skv[16] = {skA.x,skA.y,skA.z,skA.w, skB.x,skB.y,skB.z,skB.w,
                               skC.x,skC.y,skC.z,skC.w, skD.x,skD.y,skD.z,skD.w};
        const int   mkv[16] = {mA.x,mA.y,mA.z,mA.w, mB.x,mB.y,mB.z,mB.w,
                               mC.x,mC.y,mC.z,mC.w, mD.x,mD.y,mD.z,mD.w};
#pragma unroll
        for (int j = 0; j < 16; ++j) {
            float s = sqv + skv[j];
            s = (s >= 0.0f) ? s : NEG_SLOPE * s;
            pf[j] = (mkv[j] != 0) ? __expf(s - mv) : 0.0f;
        }
        const v16bf A = pack16(pf);

#pragma unroll
        for (int t = 0; t < 4; ++t) {
            const int dloc = t * 16 + l16;
            const v16bf Bm = load_v16bf(&vt_s[p][dloc * 40 + kb],
                                        &vt_s[p][dloc * 40 + kb + 16]);
            acc[t] = __builtin_amdgcn_wmma_f32_16x16x32_bf16(
                false, A, false, Bm, (short)0, acc[t], false, false);
        }
        __syncthreads();
    }

    // epilogue: normalize by Z and store (C row = vgpr + 8*(lane>=16))
#pragma unroll
    for (int j = 0; j < 8; ++j) {
        const int r = wrow + j + (hi ? 8 : 0);
        const float zz = zrow[(size_t)bh * LQ + r];
        const float sc = (zz > 0.0f) ? 1.0f / zz : 0.0f;
#pragma unroll
        for (int t = 0; t < 4; ++t) {
            const int col = h * HD + t * 16 + l16;
            out[((size_t)b * LQ + r) * Dm + col] = acc[t][j] * sc;
        }
    }
}

// ---------------------------------------------------------------------------
extern "C" void kernel_launch(void* const* d_in, const int* in_sizes, int n_in,
                              void* d_out, int out_size, void* d_ws, size_t ws_size,
                              hipStream_t stream) {
    const float* query = (const float*)d_in[0];
    const float* keyt  = (const float*)d_in[1];
    const float* value = (const float*)d_in[2];
    const int*   mask  = (const int*)d_in[3];
    const float* W     = (const float*)d_in[4];
    const float* a     = (const float*)d_in[5];
    float* out = (float*)d_out;

    float* ws = (float*)d_ws;
    const size_t BLD = (size_t)Bn * LQ * Dm;     // 2,097,152
    const size_t BHL = (size_t)Bn * Hh * LQ;     // 32,768
    const size_t WSZ = (size_t)Dm * Dm;          // 262,144
    float* Qp   = ws;
    float* Kp   = ws + BLD;
    float* Vp   = ws + 2 * BLD;
    float* sqp  = ws + 3 * BLD;
    float* skp  = sqp + BHL;
    float* mrow = skp + BHL;
    float* zrow = mrow + BHL;
    bf16raw* bw  = (bf16raw*)(zrow + BHL);       // 16B-aligned
    bf16raw* Xbq = bw;
    bf16raw* Xbk = Xbq + BLD;
    bf16raw* Xbv = Xbk + BLD;
    bf16raw* Wb  = Xbv + BLD;
    bf16raw* Vtg = Wb + WSZ;                     // [b,h,d,k] bf16

    // 0) convert inputs to bf16
    cvt_kernel<<<(int)(BLD / 1024), 256, 0, stream>>>(query, Xbq, (int)BLD);
    cvt_kernel<<<(int)(BLD / 1024), 256, 0, stream>>>(keyt,  Xbk, (int)BLD);
    cvt_kernel<<<(int)(BLD / 1024), 256, 0, stream>>>(value, Xbv, (int)BLD);
    cvt_kernel<<<(int)(WSZ / 1024), 256, 0, stream>>>(W,     Wb,  (int)WSZ);

    // 1) projections (WMMA)
    const dim3 gProj((Bn * LQ) / 128, Dm / 64);  // (32, 8)
    proj_kernel<<<gProj, 256, 0, stream>>>(Xbq, Wb, Qp);
    proj_kernel<<<gProj, 256, 0, stream>>>(Xbk, Wb, Kp);
    proj_kernel<<<gProj, 256, 0, stream>>>(Xbv, Wb, Vp);

    // 2) attention row/col scores
    sqsk_kernel<<<(2 * Bn * LQ * Hh) / 256, 256, 0, stream>>>(Qp, Kp, a, sqp, skp);

    // 3a) softmax stats
    stats_kernel<<<(Bn * Hh * LQ) / 8, 256, 0, stream>>>(sqp, skp, mask, mrow, zrow);

    // 3.5) V -> bf16 transposed
    vtrans_kernel<<<dim3(Bn * Hh, LK / 32), 256, 0, stream>>>(Vp, Vtg);

    // 3b) attention x V (WMMA)
    const dim3 gAttn(Bn * Hh, LQ / 128);         // (32, 8)
    attn_kernel<<<gAttn, 256, 0, stream>>>(Vtg, sqp, skp, mask, mrow, zrow, out);
}
